// Attention_42288247996512
// MI455X (gfx1250) — compile-verified, matrix operand order checked by reading, not run on hardware
//
#include <hip/hip_runtime.h>
#include <hip/hip_bf16.h>
#include <stdint.h>

// Problem constants (match reference)
#define BB 4
#define SSQ 2048
#define DMODEL 1024
#define DATTN 128

typedef __attribute__((ext_vector_type(16))) __bf16 bf16x16;
typedef __attribute__((ext_vector_type(8)))  float  f32x8;

union Frag16 { bf16x16 v; uint4 u[2]; };

__device__ __forceinline__ f32x8 zero_f32x8() {
  f32x8 z = {0.f, 0.f, 0.f, 0.f, 0.f, 0.f, 0.f, 0.f};
  return z;
}

// Load one 16x32 (A) or 32x16-as-rows-of-Bt (B) bf16 fragment.
// Per CDNA5 ISA 16-bit A layout: lane group g=lane/16; dwords 0-3 hold
// K = kbase+8g .. +7, dwords 4-7 hold K = kbase+16+8g .. +7. Two b128 loads.
__device__ __forceinline__ bf16x16 load_frag(const __bf16* __restrict__ base,
                                             int ld, int row, int kbase, int g) {
  Frag16 f;
  const __bf16* p = base + (size_t)row * (size_t)ld + (size_t)(kbase + 8 * g);
  f.u[0] = *(const uint4*)(p);
  f.u[1] = *(const uint4*)(p + 16);
  return f.v;
}

__device__ __forceinline__ f32x8 wmma_bf16(bf16x16 a, bf16x16 b, f32x8 c) {
  return __builtin_amdgcn_wmma_f32_16x16x32_bf16(false, a, false, b,
                                                 (short)0, c, false, false);
}

// QK^T 16x16 logits tile over K=128: two independent WMMA chains of depth 2,
// combined at the end (breaks the 4-deep serial accumulator dependency).
__device__ __forceinline__ f32x8 qk_tile(const bf16x16 qf[4],
                                         const __bf16* __restrict__ Kmat,
                                         int krow, int g) {
  bf16x16 k0 = load_frag(Kmat, DATTN, krow, 0, g);
  bf16x16 k1 = load_frag(Kmat, DATTN, krow, 32, g);
  bf16x16 k2 = load_frag(Kmat, DATTN, krow, 64, g);
  bf16x16 k3 = load_frag(Kmat, DATTN, krow, 96, g);
  f32x8 lg0 = wmma_bf16(qf[0], k0, zero_f32x8());
  f32x8 lg1 = wmma_bf16(qf[2], k2, zero_f32x8());
  lg0 = wmma_bf16(qf[1], k1, lg0);
  lg1 = wmma_bf16(qf[3], k3, lg1);
  return lg0 + lg1;
}

// ---------------- f32 -> bf16 conversion ----------------
__global__ void cvt_f32_bf16(const float* __restrict__ in,
                             __bf16* __restrict__ out, int n) {
  int stride = gridDim.x * blockDim.x;
  for (int i = blockIdx.x * blockDim.x + threadIdx.x; i < n; i += stride)
    out[i] = (__bf16)in[i];
}

// ---------------- generic projection GEMM: C[M,N] = oscale*(A @ W^T) -------
// M fixed = BB*SSQ = 8192. One wave: one 16-row M tile x NT 16-col N tiles.
template <int N, int K, bool TRANS_OUT>
__global__ __launch_bounds__(256) void gemm_bf16(const __bf16* __restrict__ A,
                                                 const __bf16* __restrict__ W,
                                                 __bf16* __restrict__ C,
                                                 float oscale) {
  constexpr int NT = 4;
  constexpr int SLOTS = N / (16 * NT);
  int wave = (blockIdx.x * blockDim.x + threadIdx.x) >> 5;
  int lane = threadIdx.x & 31;
  int g = lane >> 4, n16 = lane & 15;
  int m_tile = wave / SLOTS, slot = wave % SLOTS;
  int arow = m_tile * 16 + n16;

  f32x8 acc[NT];
#pragma unroll
  for (int j = 0; j < NT; ++j) acc[j] = zero_f32x8();

  for (int kt = 0; kt < K; kt += 32) {
    bf16x16 af = load_frag(A, K, arow, kt, g);
#pragma unroll
    for (int j = 0; j < NT; ++j) {
      int brow = (slot * NT + j) * 16 + n16;  // row of W = column of B
      bf16x16 bf = load_frag(W, K, brow, kt, g);
      acc[j] = wmma_bf16(af, bf, acc[j]);     // independent chains per j
    }
  }
  // C fragment: vgpr i, lanes 0-15 -> row i, lanes 16-31 -> row i+8; col = n16
#pragma unroll
  for (int j = 0; j < NT; ++j) {
#pragma unroll
    for (int i = 0; i < 8; ++i) {
      int row = m_tile * 16 + i + 8 * g;           // global row in [0, 8192)
      int col = (slot * NT + j) * 16 + n16;        // column in [0, N)
      float v = acc[j][i] * oscale;
      if (TRANS_OUT) {
        // store v transposed per batch: vT[b][col][s_local]
        int b = row >> 11, sl = row & (SSQ - 1);
        C[((size_t)(b * DMODEL + col)) * SSQ + sl] = (__bf16)v;
      } else {
        C[(size_t)row * N + col] = (__bf16)v;
      }
    }
  }
}

// ---------------- masked softmax -> unnormalized P (bf16) + row sums -------
// One wave per (batch, 16-row tile). Q is pre-scaled by 1/sqrt(128).
// Sub-diagonal tiles use additive -inf biases (branchless, no compares);
// only the single diagonal tile evaluates the full keep predicate, and its
// masked logits are cached in registers so sweep 2 skips it entirely.
// Normalization (1/rowsum) is deferred into the PV GEMM via rsum[].
__global__ __launch_bounds__(256) void attn_softmax(
    const __bf16* __restrict__ Q, const __bf16* __restrict__ Kmat,
    const int* __restrict__ mask, __bf16* __restrict__ P,
    float* __restrict__ rsum) {
  const float NEG_INF = -__builtin_inff();
  int wave = (blockIdx.x * blockDim.x + threadIdx.x) >> 5;
  int lane = threadIdx.x & 31;
  int g = lane >> 4, n16 = lane & 15;
  int b = wave / (SSQ / 16);
  int s_tile = wave % (SSQ / 16);
  int s_base = s_tile * 16;

  // Q fragments for the whole 16x128 row block (4 chunks of K=32)
  bf16x16 qf[4];
#pragma unroll
  for (int c = 0; c < 4; ++c)
    qf[c] = load_frag(Q, DATTN, b * SSQ + s_base + n16, c * 32, g);

  int srow[8];
  float rowbias[8];
  bool msksb[8];
#pragma unroll
  for (int i = 0; i < 8; ++i) {
    srow[i] = s_base + i + 8 * g;
    int m = mask[b * SSQ + srow[i]];
    msksb[i] = (m != 0);
    rowbias[i] = msksb[i] ? 0.f : NEG_INF;
  }

  float pm[8], diagval[8];
#pragma unroll
  for (int i = 0; i < 8; ++i) pm[i] = NEG_INF;

  // ---- sweep 1a: sub-diagonal tiles, additive bias masking ----
  for (int tt = 0; tt < s_tile; ++tt) {
    int t_col = tt * 16 + n16;
    float colbias = (mask[b * SSQ + t_col] != 0) ? 0.f : NEG_INF;
    f32x8 lg = qk_tile(qf, Kmat, b * SSQ + t_col, g);
#pragma unroll
    for (int i = 0; i < 8; ++i)
      pm[i] = fmaxf(pm[i], lg[i] + (colbias + rowbias[i]));
  }
  // ---- sweep 1b: diagonal tile, full predicate (bitwise, branchless) ----
  {
    int t_col = s_base + n16;
    bool msktb = (mask[b * SSQ + t_col] != 0);
    f32x8 lg = qk_tile(qf, Kmat, b * SSQ + t_col, g);
#pragma unroll
    for (int i = 0; i < 8; ++i) {
      bool keep = (t_col <= srow[i]) &
                  ((msksb[i] & msktb) | (t_col == srow[i]));
      float val = keep ? lg[i] : NEG_INF;
      diagval[i] = val;
      pm[i] = fmaxf(pm[i], val);
    }
  }
  // one cross-lane max reduction per wave (rows live in 16-lane groups)
  float rm[8];
#pragma unroll
  for (int i = 0; i < 8; ++i) {
    float t = pm[i];
#pragma unroll
    for (int m = 1; m <= 8; m <<= 1) t = fmaxf(t, __shfl_xor(t, m, 32));
    rm[i] = t;  // finite: diagonal element is always kept
  }

  // ---- sweep 2: exp(val - rm), per-lane partial sums, store bf16 P ----
  float ps[8];
#pragma unroll
  for (int i = 0; i < 8; ++i) ps[i] = 0.f;

  for (int tt = 0; tt < s_tile; ++tt) {
    int t_col = tt * 16 + n16;
    float colbias = (mask[b * SSQ + t_col] != 0) ? 0.f : NEG_INF;
    f32x8 lg = qk_tile(qf, Kmat, b * SSQ + t_col, g);
#pragma unroll
    for (int i = 0; i < 8; ++i) {
      float val = lg[i] + (colbias + rowbias[i]);
      float p = __expf(val - rm[i]);  // exp(-inf) == 0, rm finite
      ps[i] += p;
      P[((size_t)b * SSQ + srow[i]) * SSQ + t_col] = (__bf16)p;
    }
  }
  // diagonal tile from cached logits (no recompute)
#pragma unroll
  for (int i = 0; i < 8; ++i) {
    float p = __expf(diagval[i] - rm[i]);
    ps[i] += p;
    P[((size_t)b * SSQ + srow[i]) * SSQ + s_base + n16] = (__bf16)p;
  }
  // Pad one extra zero tile so PV GEMM can always use K chunks of 32.
  if ((s_tile & 1) == 0 && s_tile + 1 < SSQ / 16) {
#pragma unroll
    for (int i = 0; i < 8; ++i)
      P[((size_t)b * SSQ + srow[i]) * SSQ + s_base + 16 + n16] = (__bf16)0.f;
  }
  // one cross-lane sum reduction per wave; lane 0 of each group stores
#pragma unroll
  for (int i = 0; i < 8; ++i) {
    float t = ps[i];
#pragma unroll
    for (int m = 1; m <= 8; m <<= 1) t += __shfl_xor(t, m, 32);
    if (n16 == 0) rsum[(size_t)b * SSQ + srow[i]] = t;
  }
}

// ---------------- PV GEMM: out[b][s][e] = (1/rsum[s]) * sum_t P*vT ---------
__global__ __launch_bounds__(256) void pv_gemm(const __bf16* __restrict__ P,
                                               const __bf16* __restrict__ VT,
                                               const float* __restrict__ rsum,
                                               float* __restrict__ out) {
  constexpr int NT = 4;
  constexpr int SLOTS = DMODEL / (16 * NT);  // 16
  int wave = (blockIdx.x * blockDim.x + threadIdx.x) >> 5;
  int lane = threadIdx.x & 31;
  int g = lane >> 4, n16 = lane & 15;
  int b = wave / ((SSQ / 16) * SLOTS);
  int rem = wave % ((SSQ / 16) * SLOTS);
  int s_tile = rem / SLOTS, slot = rem % SLOTS;

  const __bf16* Pb = P + (size_t)b * SSQ * SSQ;
  const __bf16* Vb = VT + (size_t)b * DMODEL * SSQ;
  int arow = s_tile * 16 + n16;

  f32x8 acc[NT];
#pragma unroll
  for (int j = 0; j < NT; ++j) acc[j] = zero_f32x8();

  int kchunks = (s_tile + 2) >> 1;  // causal: t <= s_base+15, padded to 32
  for (int kc = 0; kc < kchunks; ++kc) {
    bf16x16 af = load_frag(Pb, SSQ, arow, kc * 32, g);
#pragma unroll
    for (int j = 0; j < NT; ++j) {
      int erow = (slot * NT + j) * 16 + n16;  // vT row = output column e
      bf16x16 bf = load_frag(Vb, SSQ, erow, kc * 32, g);
      acc[j] = wmma_bf16(af, bf, acc[j]);     // independent chains per j
    }
  }
  // deferred softmax normalization: multiply by 1/rowsum
  float rinv[8];
#pragma unroll
  for (int i = 0; i < 8; ++i)
    rinv[i] = 1.0f / rsum[(size_t)b * SSQ + s_tile * 16 + i + 8 * g];

#pragma unroll
  for (int j = 0; j < NT; ++j) {
#pragma unroll
    for (int i = 0; i < 8; ++i) {
      int s = s_tile * 16 + i + 8 * g;
      int e = (slot * NT + j) * 16 + n16;
      out[((size_t)b * SSQ + s) * DMODEL + e] = acc[j][i] * rinv[i];
    }
  }
}

// --------------------------------------------------------------------------
extern "C" void kernel_launch(void* const* d_in, const int* in_sizes, int n_in,
                              void* d_out, int out_size, void* d_ws,
                              size_t ws_size, hipStream_t stream) {
  const float* x     = (const float*)d_in[0];  // [B,S,DMODEL]
  const float* cross = (const float*)d_in[1];  // [B,S,DMODEL]
  const float* Wq    = (const float*)d_in[2];  // [DATTN,DMODEL]
  const float* Wk    = (const float*)d_in[3];  // [DATTN,DMODEL]
  const float* Wv    = (const float*)d_in[4];  // [DMODEL,DMODEL]
  const int*   mask  = (const int*)d_in[5];    // [B,S]
  float* out = (float*)d_out;

  const size_t M = (size_t)BB * SSQ;  // 8192
  char* ws = (char*)d_ws;
  __bf16* xb  = (__bf16*)ws;                 ws += M * DMODEL * 2;        // 16 MiB
  __bf16* cb  = (__bf16*)ws;                 ws += M * DMODEL * 2;        // 16 MiB
  __bf16* wqb = (__bf16*)ws;                 ws += (size_t)DATTN * DMODEL * 2;
  __bf16* wkb = (__bf16*)ws;                 ws += (size_t)DATTN * DMODEL * 2;
  __bf16* wvb = (__bf16*)ws;                 ws += (size_t)DMODEL * DMODEL * 2;
  __bf16* qb  = (__bf16*)ws;                 ws += M * DATTN * 2;         // 2 MiB
  __bf16* kb  = (__bf16*)ws;                 ws += M * DATTN * 2;         // 2 MiB
  __bf16* vtb = (__bf16*)ws;                 ws += M * DMODEL * 2;        // 16 MiB
  __bf16* pb  = (__bf16*)ws;                 ws += (size_t)BB * SSQ * SSQ * 2; // 32 MiB
  float*  rsm = (float*)ws;                  ws += M * 4;                 // 32 KiB
  (void)ws_size; (void)in_sizes; (void)n_in; (void)out_size;

  // 1) f32 -> bf16 conversions
  int nx = (int)(M * DMODEL);
  cvt_f32_bf16<<<2048, 256, 0, stream>>>(x, xb, nx);
  cvt_f32_bf16<<<2048, 256, 0, stream>>>(cross, cb, nx);
  cvt_f32_bf16<<<128, 256, 0, stream>>>(Wq, wqb, DATTN * DMODEL);
  cvt_f32_bf16<<<128, 256, 0, stream>>>(Wk, wkb, DATTN * DMODEL);
  cvt_f32_bf16<<<512, 256, 0, stream>>>(Wv, wvb, DMODEL * DMODEL);

  // 2) q = (x @ Wq^T)/sqrt(128), k = cross @ Wk^T
  gemm_bf16<DATTN, DMODEL, false><<<128, 256, 0, stream>>>(
      xb, wqb, qb, 0.08838834764831845f);
  gemm_bf16<DATTN, DMODEL, false><<<128, 256, 0, stream>>>(cb, wkb, kb, 1.0f);

  // 3) v = x @ Wv^T, stored transposed per batch (8192 waves -> 1024 blocks)
  gemm_bf16<DMODEL, DMODEL, true><<<1024, 256, 0, stream>>>(xb, wvb, vtb, 1.0f);

  // 4) masked softmax -> unnormalized P + row sums (512 waves -> 64 blocks)
  attn_softmax<<<64, 256, 0, stream>>>(qb, kb, mask, pb, rsm);

  // 5) out = (P @ v) * (1/rowsum)  (8192 waves -> 1024 blocks)
  pv_gemm<<<1024, 256, 0, stream>>>(pb, vtb, rsm, out);
}
